// NewDAGExecutor_24369644437717
// MI455X (gfx1250) — matrix operand order repaired and checked by reading, not compiled, not run on hardware
//
#include <hip/hip_runtime.h>
#include <math.h>

// ---------------------------------------------------------------------------
// NewDAGExecutor fused kernel for gfx1250 (MI455X)
//   Phase 0 (prepack): W_init/W_op/W_gate -> one padded 176x2048 f16 matrix
//   Phase 1 (fused):   f16 WMMA GEMM (f32 accum) + per-token DAG recurrence
// Software-pipelined: next K-step's A/B fragments are issued before the
// current step's WMMAs consume their registers; next chunk is prefetched.
// ---------------------------------------------------------------------------

typedef __attribute__((ext_vector_type(16))) _Float16 v16h;
typedef __attribute__((ext_vector_type(8)))  _Float16 v8h;
typedef __attribute__((ext_vector_type(4)))  _Float16 v4h;
typedef __attribute__((ext_vector_type(8)))  float    v8f;

#define H        2048      // hidden size (K)
#define NCOLS    168       // 32 init + 128 op + 8 gate
#define NCOLSP   176       // padded to 11 * 16
#define NWAVES   11        // one wave per 16-col tile
#define THREADS  (NWAVES * 32)   // 352
#define MB       32        // tokens per workgroup (2 M-tiles of 16)
#define KC       512       // K-chunk staged in LDS
#define KSTEPS   (KC / 32) // WMMA K-steps per chunk
#define ASTRIDE  520       // padded f16 row stride in LDS (bank-conflict free)
#define OSTRIDE  177       // padded f32 out row stride in LDS
#define NTOK     16384     // B * T

#define SHUF16(lo, hi) __builtin_shufflevector(lo, hi, \
        0,1,2,3,4,5,6,7,8,9,10,11,12,13,14,15)

// --------------------------- weight prepack --------------------------------
__global__ void prepack_weights(const float* __restrict__ Wi, const float* __restrict__ bi,
                                const float* __restrict__ Wo, const float* __restrict__ bo,
                                const float* __restrict__ Wg, const float* __restrict__ bg,
                                _Float16* __restrict__ W16, float* __restrict__ b176) {
  int idx = blockIdx.x * 256 + threadIdx.x;
  if (idx < NCOLSP * H) {
    int n = idx >> 11;          // row (output column)
    int k = idx & (H - 1);
    float v = 0.0f;
    if      (n < 32)  v = Wi[n * H + k];
    else if (n < 160) v = Wo[(n - 32) * H + k];
    else if (n < 168) v = Wg[(n - 160) * H + k];
    W16[idx] = (_Float16)v;
  }
  if (blockIdx.x == 0 && threadIdx.x < NCOLSP) {
    int n = threadIdx.x;
    float v = 0.0f;
    if      (n < 32)  v = bi[n];
    else if (n < 160) v = bo[n - 32];
    else if (n < 168) v = bg[n - 160];
    b176[n] = v;
  }
}

// --------------------------- fused GEMM + DAG ------------------------------
__global__ __launch_bounds__(THREADS, 1)
void dag_exec_kernel(const float* __restrict__ hidden,
                     const _Float16* __restrict__ W16,
                     const float* __restrict__ b176,
                     float* __restrict__ out) {
  __shared__ _Float16 ldsA[MB * ASTRIDE];     // 33,280 B  (f16 hidden chunk)
  __shared__ float    ldsOut[MB * OSTRIDE];   // 22,656 B  (GEMM results)

  const int tid  = threadIdx.x;
  const int wave = tid >> 5;
  const int lane = tid & 31;
  const int lh   = lane >> 4;      // lane half (0/1)
  const int ln   = lane & 15;
  const int n    = wave * 16 + ln; // my output column (0..175)
  const int tokBase = blockIdx.x * MB;

  v8f acc0 = {};                   // tokens 0..15 of this block
  v8f acc1 = {};                   // tokens 16..31
  const _Float16* wrow = W16 + (size_t)n * H + lh * 16;

  for (int ck = 0; ck < H; ck += KC) {
    // ---- stage hidden f32 -> f16 into LDS (cooperative, coalesced) ----
    for (int i = tid; i < MB * (KC / 4); i += THREADS) {
      int tok = i >> 7;                 // KC/4 = 128 float4 per token
      int kq  = (i & 127) << 2;
      const float4 v = *reinterpret_cast<const float4*>(
          hidden + (size_t)(tokBase + tok) * H + ck + kq);
      v4h h;
      h[0] = (_Float16)v.x; h[1] = (_Float16)v.y;
      h[2] = (_Float16)v.z; h[3] = (_Float16)v.w;
      *reinterpret_cast<v4h*>(&ldsA[tok * ASTRIDE + kq]) = h;
    }
    __syncthreads();

    // ---- prefetch next chunk of hidden into cache during compute ----
    if (ck + KC < H) {
      for (int i = tid; i < MB * (KC / 4); i += THREADS) {
        int tok = i >> 7;
        int kq  = (i & 127) << 2;
        __builtin_prefetch(hidden + (size_t)(tokBase + tok) * H + ck + KC + kq, 0, 0);
      }
    }

    // ---- software-pipelined WMMA K-steps over this chunk ----
    const _Float16* abase0 = &ldsA[ln * ASTRIDE + lh * 16];         // tile 0
    const _Float16* abase1 = &ldsA[(16 + ln) * ASTRIDE + lh * 16];  // tile 1
    const _Float16* bbase  = wrow + ck;

    v16h bcur = *reinterpret_cast<const v16h*>(bbase);
    v8h  a0lo = *reinterpret_cast<const v8h*>(abase0);
    v8h  a0hi = *reinterpret_cast<const v8h*>(abase0 + 8);
    v8h  a1lo = *reinterpret_cast<const v8h*>(abase1);
    v8h  a1hi = *reinterpret_cast<const v8h*>(abase1 + 8);

    #pragma unroll
    for (int ks = 0; ks < KSTEPS; ++ks) {
      // issue next iteration's loads before consuming current registers
      v16h bnxt = bcur;
      v8h  n0lo = a0lo, n0hi = a0hi, n1lo = a1lo, n1hi = a1hi;
      if (ks + 1 < KSTEPS) {
        const int o = (ks + 1) * 32;
        bnxt = *reinterpret_cast<const v16h*>(bbase + o);
        n0lo = *reinterpret_cast<const v8h*>(abase0 + o);
        n0hi = *reinterpret_cast<const v8h*>(abase0 + o + 8);
        n1lo = *reinterpret_cast<const v8h*>(abase1 + o);
        n1hi = *reinterpret_cast<const v8h*>(abase1 + o + 8);
      }
      v16h a0 = SHUF16(a0lo, a0hi);
      v16h a1 = SHUF16(a1lo, a1hi);
      acc0 = __builtin_amdgcn_wmma_f32_16x16x32_f16(
               false, a0, false, bcur, (short)0, acc0, false, false);
      acc1 = __builtin_amdgcn_wmma_f32_16x16x32_f16(
               false, a1, false, bcur, (short)0, acc1, false, false);
      bcur = bnxt;
      a0lo = n0lo; a0hi = n0hi; a1lo = n1lo; a1hi = n1hi;
    }
    __syncthreads();
  }

  // ---- bias add + write accumulators to LDS (C/D layout: row r+8*lh, col ln)
  const float bv = b176[n];
  #pragma unroll
  for (int r = 0; r < 8; ++r) {
    ldsOut[(r + 8 * lh) * OSTRIDE + n]      = acc0[r] + bv;
    ldsOut[(16 + r + 8 * lh) * OSTRIDE + n] = acc1[r] + bv;
  }
  __syncthreads();

  // ---- per-token DAG recurrence (one lane per token, fully unrolled) ----
  if (tid < MB) {
    const float* myrow = &ldsOut[tid * OSTRIDE];
    float Vm[16], Vs[16], Lg[16], Sg[16];
    #pragma unroll
    for (int j = 0; j < 16; ++j) {
      float ir = myrow[j];
      float is = myrow[16 + j];
      Vm[j] = fabsf(ir);
      Vs[j] = tanhf(is);
      Lg[j] = logf(fmaxf(Vm[j], 1e-12f));
      Sg[j] = Vs[j] * Vm[j];
    }
    #pragma unroll
    for (int step = 0; step < 8; ++step) {
      const int valid = 8 + step;
      const float G = 1.0f / (1.0f + expf(-myrow[160 + step]));
      float R = 0.0f, Pa = 1.0f, Ps = 1.0f;
      #pragma unroll
      for (int j = 0; j < 16; ++j) {
        Ps *= Vs[j];                              // sign product over all slots
        if (j < valid) {                          // masked op row
          float o = myrow[32 + step * 16 + j];
          R += o * (Lg[j] * (1.0f - G) + Sg[j] * G);
          Pa *= fabsf(o) + 1.0f;
        }
      }
      const float P        = Ps * Pa;
      const float lin_sign = tanhf(R * 1.0e4f);   // tanh(R / SIGN_TEMP)
      const float log_sign = tanhf(P * 1.0e4f);
      const float vs = G * lin_sign + (1.0f - G) * log_sign;
      const float vm = G * fabsf(R) + (1.0f - G) * expf(fminf(R, 23.026f));
      Vm[valid] = vm;
      Vs[valid] = vs;
      Lg[valid] = logf(fmaxf(vm, 1e-12f));
      Sg[valid] = vs * vm;
    }
    out[tokBase + tid] = Vs[15] * Vm[15];
  }
}

// ------------------------------- launcher ----------------------------------
extern "C" void kernel_launch(void* const* d_in, const int* in_sizes, int n_in,
                              void* d_out, int out_size, void* d_ws, size_t ws_size,
                              hipStream_t stream) {
  (void)in_sizes; (void)n_in; (void)out_size; (void)ws_size;
  const float* hidden = (const float*)d_in[0];
  const float* W_init = (const float*)d_in[1];
  const float* b_init = (const float*)d_in[2];
  const float* W_op   = (const float*)d_in[3];
  const float* b_op   = (const float*)d_in[4];
  const float* W_gate = (const float*)d_in[5];
  const float* b_gate = (const float*)d_in[6];
  float* out = (float*)d_out;

  _Float16* W16 = reinterpret_cast<_Float16*>(d_ws);
  float* b176   = reinterpret_cast<float*>((char*)d_ws + (size_t)NCOLSP * H * sizeof(_Float16));

  const int prepack_blocks = (NCOLSP * H + 255) / 256;
  prepack_weights<<<prepack_blocks, 256, 0, stream>>>(W_init, b_init, W_op, b_op,
                                                      W_gate, b_gate, W16, b176);

  dag_exec_kernel<<<NTOK / MB, THREADS, 0, stream>>>(hidden, W16, b176, out);
}
